// TrulyNeuralMMUv2_18975165514017
// MI455X (gfx1250) — matrix operand order repaired and checked by reading, not compiled, not run on hardware
//
#include <hip/hip_runtime.h>
#include <hip/hip_bf16.h>
#include <math.h>

typedef __attribute__((ext_vector_type(16))) _Float16 v16h;
typedef __attribute__((ext_vector_type(8)))  _Float16 v8h;
typedef __attribute__((ext_vector_type(8)))  float    v8f;

#define BATCH 32768
#define PAGES 4096
#define KD    64
#define PB    20

// ---- workspace byte offsets ----
// q16    : BATCH*64 f16 = 4194304  @ 0
// vpb2   : BATCH*32 f16 = 2097152  @ 4194304
// xterm  : BATCH    f32 = 131072   @ 6291456
// keys16 : PAGES*64 f16 = 524288   @ 6422528
// svirt2 : PAGES*32 f16 = 262144   @ 6946816
// vt16   : 32*PAGES f16 = 262144   @ 7208960
// sterm  : PAGES    f32 = 16384    @ 7471104   (total ~7.3 MB)

static __device__ __forceinline__ v8h LD8(const _Float16* p) { return *(const v8h*)p; }

static __device__ __forceinline__ v8h zero8() {
    v8h z;
#pragma unroll
    for (int i = 0; i < 8; ++i) z[i] = (_Float16)0.f;
    return z;
}
static __device__ __forceinline__ v8f zerof8() {
    v8f z;
#pragma unroll
    for (int i = 0; i < 8; ++i) z[i] = 0.f;
    return z;
}
static __device__ __forceinline__ v16h combine(v8h lo, v8h hi) {
    v16h r;
#pragma unroll
    for (int i = 0; i < 8; ++i) { r[i] = lo[i]; r[i + 8] = hi[i]; }
    return r;
}
static __device__ __forceinline__ float sigmoidf_(float x) {
    return 1.0f / (1.0f + __expf(-x));
}

// ---------------- Phase 0: page-side preprocessing ----------------
__global__ void page_preprocess(const float* __restrict__ page_keys,
                                const float* __restrict__ virtual_pages,
                                const float* __restrict__ physical_pages,
                                const float* __restrict__ permissions,
                                const float* __restrict__ temperature,
                                _Float16* __restrict__ keys16,
                                _Float16* __restrict__ svirt2,
                                _Float16* __restrict__ vt16,
                                float* __restrict__ sterm)
{
    int p = blockIdx.x * blockDim.x + threadIdx.x;
    if (p >= PAGES) return;
    float invT = 1.0f / fmaxf(fabsf(temperature[0]), 0.1f);

#pragma unroll 8
    for (int c = 0; c < KD; ++c) keys16[p * KD + c] = (_Float16)page_keys[p * KD + c];

    float s2 = 0.f;
#pragma unroll
    for (int c = 0; c < 32; ++c) {
        float v = 0.f;
        if (c < PB) { float sv = sigmoidf_(virtual_pages[p * PB + c]); s2 += sv * sv; v = 2.f * sv; }
        svirt2[p * 32 + c] = (_Float16)v;   // pre-scaled x2 so WMMA gives 4*cross
    }
    sterm[p] = -2.f * s2 * invT;

#pragma unroll
    for (int c = 0; c < 32; ++c) {
        float v = 0.f;
        if (c < PB)      v = sigmoidf_(physical_pages[p * PB + c]);
        else if (c < 24) v = sigmoidf_(permissions[p * 4 + (c - PB)]);
        vt16[c * PAGES + p] = (_Float16)v;  // transposed [32][PAGES], rows 24..31 zero
    }
}

// ---------------- Phase 1: query MLP (20 -> 128 GELU -> 64) ----------------
__global__ void query_encode(const float* __restrict__ vpb,
                             const float* __restrict__ W1, const float* __restrict__ b1,
                             const float* __restrict__ W2, const float* __restrict__ b2,
                             const float* __restrict__ temperature,
                             _Float16* __restrict__ q16,
                             _Float16* __restrict__ vpb2,
                             float* __restrict__ xterm)
{
    int r = blockIdx.x * blockDim.x + threadIdx.x;
    if (r >= BATCH) return;
    float invT = 1.0f / fmaxf(fabsf(temperature[0]), 0.1f);

    float x[PB]; float x2 = 0.f;
#pragma unroll
    for (int c = 0; c < PB; ++c) { x[c] = vpb[r * PB + c]; x2 += x[c] * x[c]; }

    float q[KD];
#pragma unroll
    for (int k = 0; k < KD; ++k) q[k] = b2[k];

    for (int j = 0; j < 2 * KD; ++j) {
        float z = b1[j];
#pragma unroll
        for (int c = 0; c < PB; ++c) z += x[c] * W1[c * (2 * KD) + j];
        float h = 0.5f * z * (1.0f + erff(z * 0.70710678118654752f));  // exact GELU
#pragma unroll
        for (int k = 0; k < KD; ++k) q[k] += h * W2[j * KD + k];
    }
#pragma unroll
    for (int k = 0; k < KD; ++k) q16[r * KD + k] = (_Float16)q[k];
#pragma unroll
    for (int c = 0; c < 32; ++c) vpb2[r * 32 + c] = (_Float16)((c < PB) ? 2.f * x[c] : 0.f);
    xterm[r] = -2.f * x2 * invT;
}

// ---------------- Phase 2: fused sim + softmax + attention@V ----------------
__global__ void __launch_bounds__(256, 1)
mmu_attention(const _Float16* __restrict__ q16, const _Float16* __restrict__ vpb2,
              const float* __restrict__ xterm, const _Float16* __restrict__ keys16,
              const _Float16* __restrict__ svirt2, const _Float16* __restrict__ vt16,
              const float* __restrict__ sterm, const float* __restrict__ temperature,
              float* __restrict__ out)
{
    __shared__ __align__(16) _Float16 pbuf[8][16][16];   // per-wave P-tile bounce

    const int w    = threadIdx.x >> 5;
    const int lane = threadIdx.x & 31;
    const int half = lane >> 4;      // K/M half select per ISA fragment layout
    const int n    = lane & 15;      // column (N) / A-row index
    const int tile = blockIdx.x * 8 + w;
    const int row0 = tile * 16;
    const float invT = 1.0f / fmaxf(fabsf(temperature[0]), 0.1f);

    // A fragments (resident for whole kernel): 16x32 f16 layout
    const int rowA = row0 + n;
    v16h A0 = combine(LD8(q16 + rowA * KD + half * 16),      LD8(q16 + rowA * KD + half * 16 + 8));
    v16h A1 = combine(LD8(q16 + rowA * KD + 32 + half * 16), LD8(q16 + rowA * KD + 32 + half * 16 + 8));
    v16h Ax = combine(LD8(vpb2 + rowA * 32 + half * 16),     LD8(vpb2 + rowA * 32 + half * 16 + 8));

    float xt[8];
#pragma unroll
    for (int v = 0; v < 8; ++v) xt[v] = xterm[row0 + half * 8 + v];   // C-layout row = v + 8*half

    // ---- pass 1: row max over all pages ----
    float mmax[8];
#pragma unroll
    for (int v = 0; v < 8; ++v) mmax[v] = -3.4e38f;

    for (int t = 0; t < PAGES / 16; ++t) {
        const int pg = t * 16 + n;
        const int kh = half * 16;
        if (t + 1 < PAGES / 16) __builtin_prefetch(keys16 + (t + 1) * 16 * KD + n * KD, 0, 0);
        v16h Bk0 = combine(LD8(keys16 + pg * KD + kh),      LD8(keys16 + pg * KD + kh + 8));
        v16h Bk1 = combine(LD8(keys16 + pg * KD + 32 + kh), LD8(keys16 + pg * KD + 32 + kh + 8));
        v16h Bs  = combine(LD8(svirt2 + pg * 32 + kh),      LD8(svirt2 + pg * 32 + kh + 8));
        const float st = sterm[pg];
        v8f C = zerof8();
        C = __builtin_amdgcn_wmma_f32_16x16x32_f16(false, A0, false, Bk0, (short)0, C, false, false);
        C = __builtin_amdgcn_wmma_f32_16x16x32_f16(false, A1, false, Bk1, (short)0, C, false, false);
        C = __builtin_amdgcn_wmma_f32_16x16x32_f16(false, Ax, false, Bs,  (short)0, C, false, false);
#pragma unroll
        for (int v = 0; v < 8; ++v) {
            float s = C[v] * invT + xt[v] + st;
            mmax[v] = fmaxf(mmax[v], s);
        }
    }
#pragma unroll
    for (int m = 1; m < 16; m <<= 1)
#pragma unroll
        for (int v = 0; v < 8; ++v) mmax[v] = fmaxf(mmax[v], __shfl_xor(mmax[v], m, 32));

    // ---- pass 2: exp, row-sum, and P @ V via WMMA ----
    float lsum[8];
#pragma unroll
    for (int v = 0; v < 8; ++v) lsum[v] = 0.f;
    v8f O0 = zerof8(), O1 = zerof8();

    for (int t = 0; t < PAGES / 16; ++t) {
        const int pg = t * 16 + n;
        const int kh = half * 16;
        v16h Bk0 = combine(LD8(keys16 + pg * KD + kh),      LD8(keys16 + pg * KD + kh + 8));
        v16h Bk1 = combine(LD8(keys16 + pg * KD + 32 + kh), LD8(keys16 + pg * KD + 32 + kh + 8));
        v16h Bs  = combine(LD8(svirt2 + pg * 32 + kh),      LD8(svirt2 + pg * 32 + kh + 8));
        const float st = sterm[pg];
        v8f C = zerof8();
        C = __builtin_amdgcn_wmma_f32_16x16x32_f16(false, A0, false, Bk0, (short)0, C, false, false);
        C = __builtin_amdgcn_wmma_f32_16x16x32_f16(false, A1, false, Bk1, (short)0, C, false, false);
        C = __builtin_amdgcn_wmma_f32_16x16x32_f16(false, Ax, false, Bs,  (short)0, C, false, false);

#pragma unroll
        for (int v = 0; v < 8; ++v) {
            float s = C[v] * invT + xt[v] + st;
            float p = __expf(s - mmax[v]);
            lsum[v] += p;
            pbuf[w][half * 8 + v][n] = (_Float16)p;   // C-layout -> LDS (row, page)
        }
        // same-wave RAW through LDS: order stores before cross-lane loads
        asm volatile("s_wait_dscnt 0x0" ::: "memory");

        v16h Ap = half ? combine(zero8(), zero8()) : combine(LD8(&pbuf[w][n][0]), LD8(&pbuf[w][n][8]));  // A-layout, K 0..15 in lower-half lanes, upper half = 0

        v16h Bv0, Bv1;
        if (lane < 16) {   // K rows 0..15 live in lower-half lanes; upper half = 0
            Bv0 = combine(LD8(vt16 + n * PAGES + t * 16),          LD8(vt16 + n * PAGES + t * 16 + 8));
            Bv1 = combine(LD8(vt16 + (16 + n) * PAGES + t * 16),   LD8(vt16 + (16 + n) * PAGES + t * 16 + 8));
        } else {
            Bv0 = combine(zero8(), zero8());
            Bv1 = combine(zero8(), zero8());
        }
        O0 = __builtin_amdgcn_wmma_f32_16x16x32_f16(false, Ap, false, Bv0, (short)0, O0, false, false);
        O1 = __builtin_amdgcn_wmma_f32_16x16x32_f16(false, Ap, false, Bv1, (short)0, O1, false, false);
        asm volatile("" ::: "memory");  // keep next iter's LDS stores after this iter's loads
    }
#pragma unroll
    for (int m = 1; m < 16; m <<= 1)
#pragma unroll
        for (int v = 0; v < 8; ++v) lsum[v] += __shfl_xor(lsum[v], m, 32);

    // ---- write outputs: ppb[B,20] | perm0[B] | perms3[B,3] ----
#pragma unroll
    for (int v = 0; v < 8; ++v) {
        const int row = row0 + half * 8 + v;
        const float inv = 1.0f / lsum[v];
        out[row * PB + n] = O0[v] * inv;           // cols 0..15
        const int c2 = 16 + n;
        const float val = O1[v] * inv;
        if (c2 < PB)       out[row * PB + c2] = val;                       // cols 16..19
        else if (c2 == PB) out[BATCH * PB + row] = val;                    // perm col 0
        else if (c2 < 24)  out[BATCH * (PB + 1) + row * 3 + (c2 - 21)] = val; // perm cols 1..3
    }
}

extern "C" void kernel_launch(void* const* d_in, const int* in_sizes, int n_in,
                              void* d_out, int out_size, void* d_ws, size_t ws_size,
                              hipStream_t stream)
{
    const float* vpb            = (const float*)d_in[0];
    const float* page_keys      = (const float*)d_in[1];
    const float* virtual_pages  = (const float*)d_in[2];
    const float* physical_pages = (const float*)d_in[3];
    const float* permissions    = (const float*)d_in[4];
    const float* W1             = (const float*)d_in[5];
    const float* b1             = (const float*)d_in[6];
    const float* W2             = (const float*)d_in[7];
    const float* b2             = (const float*)d_in[8];
    const float* temperature    = (const float*)d_in[9];
    float* out = (float*)d_out;

    char* ws = (char*)d_ws;
    _Float16* q16    = (_Float16*)(ws + 0);
    _Float16* vpb2   = (_Float16*)(ws + 4194304);
    float*    xterm  = (float*)   (ws + 6291456);
    _Float16* keys16 = (_Float16*)(ws + 6422528);
    _Float16* svirt2 = (_Float16*)(ws + 6946816);
    _Float16* vt16   = (_Float16*)(ws + 7208960);
    float*    sterm  = (float*)   (ws + 7471104);

    page_preprocess<<<PAGES / 256, 256, 0, stream>>>(
        page_keys, virtual_pages, physical_pages, permissions, temperature,
        keys16, svirt2, vt16, sterm);
    query_encode<<<BATCH / 256, 256, 0, stream>>>(
        vpb, W1, b1, W2, b2, temperature, q16, vpb2, xterm);
    mmu_attention<<<BATCH / 128, 256, 0, stream>>>(
        q16, vpb2, xterm, keys16, svirt2, vt16, sterm, temperature, out);
}
